// Connection_v5extend_17076789969616
// MI455X (gfx1250) — matrix operand order repaired and checked by reading, not compiled, not run on hardware
//
#include <hip/hip_runtime.h>
#include <hip/hip_bf16.h>

// ---------------------------------------------------------------------------
// Connection_v5extend for MI455X (gfx1250).
//
// Factorized into 6 GEMMs (no (B,128,128) Jacobian materialization):
//   h = x@W1^T+b1 ; a=relu(h) ; mask=h>0
//   z = a@W2^T+b2 ; s=sigmoid(z) ; c=sgn*s(1-s) ; g_inv=1/(sgn*(s+C)) ; p=v^2*c
//   w = v@W1^T ; u = p@W2
//   t1 = (mask*u)@W1 ; t2 = (mask*w)@W2^T
//   dv = (2*v*c*t2 - t1) * g_inv ; out = [v, dv]
//
// All GEMMs on V_WMMA_F32_16X16X32_F16 (f16 in, f32 accumulate); elementwise
// in f32. A pre-pass converts W1/W2 to f16 in BOTH orientations into d_ws so
// every B fragment is a contiguous 2x(128-bit) load: canonical copies are
// staged to LDS (b128 vector copies), transposed copies are read straight
// from global (L2/WGP$-resident, shared by all waves) -- no 16-wide u16
// gathers. Two waves cooperate per 16-row batch tile, split over N-tiles,
// with __syncthreads at the true cross-wave dependencies.
// ---------------------------------------------------------------------------

typedef __attribute__((ext_vector_type(16))) _Float16 v16h;
typedef __attribute__((ext_vector_type(8)))  _Float16 v8h;
typedef __attribute__((ext_vector_type(8)))  float    v8f;
typedef __attribute__((ext_vector_type(4)))  unsigned int u32x4;

#define CONSTV 0.618f

__device__ __forceinline__ v16h cat16(v8h lo, v8h hi) {
  union { v16h v; v8h h[2]; } u;
  u.h[0] = lo; u.h[1] = hi;
  return u.v;
}

// A fragment (16x32, f16), A stored row-major [16][strideK] in LDS.
// ISA 7.12.2: lane<16 -> M=lane, h[0..7]=K 0..7,  h[8..15]=K 16..23
//             lane>=16 -> M=lane-16, h[0..7]=K 8..15, h[8..15]=K 24..31
__device__ __forceinline__ v16h load_afrag(const _Float16* A, int strideK,
                                           int kt, int lane) {
  int m = lane & 15, ksel = (lane >> 4) & 1;
  const _Float16* p = A + m * strideK + kt * 32 + ksel * 8;
  return cat16(*(const v8h*)p, *(const v8h*)(p + 16));
}

// B fragment (32x16, f16) from N-major storage Bt[n][k] (contiguous in K).
// h[t] = B[kt*32 + ksel*8 + t][n] (t<8), B[kt*32 + 16 + ksel*8 + t-8][n] (t>=8),  n = lane&15, ksel = lane>>4.
// Works for LDS or global pointers (ds_load_b128 / global_load_b128 x2).
__device__ __forceinline__ v16h load_bfrag_nk(const _Float16* Bt, int strideN,
                                              int kt, int lane) {
  int n = lane & 15, ksel = (lane >> 4) & 1;
  const _Float16* p = Bt + n * strideN + kt * 32 + ksel * 8;
  return cat16(*(const v8h*)p, *(const v8h*)(p + 16));
}

__device__ __forceinline__ v8f wmma_f16(v16h a, v16h b, v8f c) {
  return __builtin_amdgcn_wmma_f32_16x16x32_f16(
      /*neg_a=*/false, a, /*neg_b=*/false, b,
      /*c_mod=*/(short)0, c, /*reuse_a=*/false, /*reuse_b=*/false);
}

// ws (f16) layout, in halves:
//   [0,     32768)  W1h   [256][128]  row-major W1     (N-major of W1^T)
//   [32768, 65536)  W2h   [128][256]  row-major W2     (N-major of W2^T)
//   [65536, 98304)  W1Th  [128][256]  W1Th[j][k]=W1[k][j] (N-major of W1)
//   [98304,131072)  W2Th  [256][128]  W2Th[n][k]=W2[k][n] (N-major of W2)
__global__ void convert_weights(const float* __restrict__ W1,
                                const float* __restrict__ W2,
                                _Float16* __restrict__ wsh) {
  int i = blockIdx.x * blockDim.x + threadIdx.x;
  if (i < 32768) {                       // W1 is 256 x 128
    float w = W1[i];
    int k = i >> 7, j = i & 127;         // element W1[k][j]
    wsh[i] = (_Float16)w;
    wsh[65536 + j * 256 + k] = (_Float16)w;
  } else if (i < 65536) {                // W2 is 128 x 256
    int idx = i - 32768;
    float w = W2[idx];
    int r = idx >> 8, n = idx & 255;     // element W2[r][n]
    wsh[i] = (_Float16)w;
    wsh[98304 + n * 128 + r] = (_Float16)w;
  }
}

// LDS layout (bytes):
//   [0,      65536)  sW1  f16 [256][128]
//   [65536, 131072)  sW2  f16 [128][256]
//   [131072,132096)  sB1  f32 [256]
//   [132096,132608)  sB2  f32 [128]
//   [132608, +2*40960) per-rowTile scratch:
//      fX [16][128] f16 (x, later p)     fV [16][128] f16 (v)
//      fA [16][256] f16 (a, later mu)    fMW [16][256] f16 (mask*w)
//      sGV[16][128][2] f32 {g_inv, v*c}
#define SMEM_TOTAL (132608 + 2 * 40960)   // 214528 B (<= 320KB WGP LDS)

__global__ void __launch_bounds__(128, 1)
conn_fused(const float* __restrict__ in, const _Float16* __restrict__ wsh,
           const float* __restrict__ b1, const float* __restrict__ b2,
           float* __restrict__ out) {
  extern __shared__ char smem[];
  _Float16* sW1 = (_Float16*)smem;                 // [256][128]
  _Float16* sW2 = sW1 + 32768;                     // [128][256]
  float*    sB1 = (float*)(smem + 131072);
  float*    sB2 = (float*)(smem + 132096);

  const _Float16* gW1T = wsh + 65536;              // [128][256] in global
  const _Float16* gW2T = wsh + 98304;              // [256][128] in global

  const int tid  = threadIdx.x;
  const int lane = tid & 31;          // wave32
  const int wave = tid >> 5;          // 0..3
  const int rowTile = wave >> 1;      // 0..1 : which 16-row batch tile
  const int sub     = wave & 1;       // 0..1 : N-tile split within pair

  char* tb = smem + 132608 + rowTile * 40960;
  _Float16* fX  = (_Float16*)tb;        // [16][128]
  _Float16* fV  = fX + 2048;            // [16][128]
  _Float16* fA  = fV + 2048;            // [16][256]
  _Float16* fMW = fA + 4096;            // [16][256]
  float*    sGV = (float*)(fMW + 4096); // [16][128][2]

  const int bBase = blockIdx.x * 32 + rowTile * 16;

  // ---- stage canonical weights (f16, preconverted): 8192 x 16B copies ----
  {
    const u32x4* src = (const u32x4*)wsh;
    u32x4* dst = (u32x4*)sW1;           // sW1 and sW2 are contiguous
    for (int i = tid; i < 8192; i += 128) dst[i] = src[i];
    for (int i = tid; i < 256; i += 128) sB1[i] = b1[i];
    if (tid < 128) sB2[tid] = b2[tid];
  }
  // ---- stage x, v (f16) for this row tile; wave-pair cooperative ----
  for (int i = sub * 32 + lane; i < 2048; i += 64) {
    int r = i >> 7, j = i & 127;
    fX[i] = (_Float16)in[(bBase + r) * 256 + j];
    fV[i] = (_Float16)in[(bBase + r) * 256 + 128 + j];
  }
  __syncthreads();

  const int col = lane & 15;
  const int hi  = (lane >> 4) & 1;    // C-tile: acc[g] = C[g + 8*hi][col]

  // ================= stage 1: h = x @ W1^T + b1 ; a = relu(h) -> fA ========
  for (int i = 0; i < 8; ++i) {
    int nt = sub * 8 + i;             // N tile of 256
    v8f acc = {};
    #pragma unroll
    for (int kt = 0; kt < 4; ++kt)    // K = 128
      acc = wmma_f16(load_afrag(fX, 128, kt, lane),
                     load_bfrag_nk(sW1 + nt * 16 * 128, 128, kt, lane), acc);
    int k = nt * 16 + col;
    float bb = sB1[k];
    #pragma unroll
    for (int g = 0; g < 8; ++g) {
      int row = g + 8 * hi;
      float h = acc[g] + bb;
      fA[row * 256 + k] = (_Float16)(h > 0.f ? h : 0.f);
    }
  }

  // ================= stage 2: w = v @ W1^T ; mw = mask*w -> fMW ============
  // (mask recovered from own-lane a values written in stage 1; no sync needed)
  for (int i = 0; i < 8; ++i) {
    int nt = sub * 8 + i;
    v8f acc = {};
    #pragma unroll
    for (int kt = 0; kt < 4; ++kt)
      acc = wmma_f16(load_afrag(fV, 128, kt, lane),
                     load_bfrag_nk(sW1 + nt * 16 * 128, 128, kt, lane), acc);
    int k = nt * 16 + col;
    #pragma unroll
    for (int g = 0; g < 8; ++g) {
      int row = g + 8 * hi;
      bool msk = (float)fA[row * 256 + k] > 0.f;
      fMW[row * 256 + k] = msk ? (_Float16)acc[g] : (_Float16)0.f;
    }
  }
  __syncthreads();

  // ======= stage 3: z = a @ W2^T + b2 ; s = sigmoid(z) ; g_inv, v*c, p =====
  for (int i = 0; i < 4; ++i) {
    int nt = sub * 4 + i;             // N tile of 128
    v8f acc = {};
    #pragma unroll
    for (int kt = 0; kt < 8; ++kt)    // K = 256
      acc = wmma_f16(load_afrag(fA, 256, kt, lane),
                     load_bfrag_nk(sW2 + nt * 16 * 256, 256, kt, lane), acc);
    int j = nt * 16 + col;
    float bb  = sB2[j];
    float sgn = (j < 4) ? -1.f : 1.f;
    #pragma unroll
    for (int g = 0; g < 8; ++g) {
      int row = g + 8 * hi;
      float z = acc[g] + bb;
      float s = 1.f / (1.f + __expf(-z));
      float c = sgn * s * (1.f - s);
      float vf = in[(bBase + row) * 256 + 128 + j];   // v in f32
      sGV[(row * 128 + j) * 2 + 0] = 1.f / (sgn * (s + CONSTV));
      sGV[(row * 128 + j) * 2 + 1] = vf * c;
      fX[row * 128 + j] = (_Float16)(vf * vf * c);    // p overwrites x
    }
  }
  __syncthreads();

  // ================= stage 4: u = p @ W2 ; mu = mask*u -> fA ===============
  // B = W2 (K-major) -> contiguous fragments from global W2T (L2-resident).
  for (int i = 0; i < 8; ++i) {
    int nt = sub * 8 + i;
    v8f acc = {};
    #pragma unroll
    for (int kt = 0; kt < 4; ++kt)
      acc = wmma_f16(load_afrag(fX, 128, kt, lane),
                     load_bfrag_nk(gW2T + nt * 16 * 128, 128, kt, lane), acc);
    int k = nt * 16 + col;
    #pragma unroll
    for (int g = 0; g < 8; ++g) {
      int row = g + 8 * hi;
      bool msk = (float)fA[row * 256 + k] > 0.f;      // read mask, then
      fA[row * 256 + k] = msk ? (_Float16)acc[g] : (_Float16)0.f;  // overwrite
    }
  }
  __syncthreads();

  // ========== stage 5: t1 = mu @ W1 ; t2 = mw @ W2^T ; dv -> out ===========
  // t1's B = W1 (K-major) -> contiguous fragments from global W1T.
  for (int i = 0; i < 4; ++i) {
    int nt = sub * 4 + i;
    v8f t1 = {}, t2 = {};
    #pragma unroll
    for (int kt = 0; kt < 8; ++kt) {
      t1 = wmma_f16(load_afrag(fA, 256, kt, lane),
                    load_bfrag_nk(gW1T + nt * 16 * 256, 256, kt, lane), t1);
      t2 = wmma_f16(load_afrag(fMW, 256, kt, lane),
                    load_bfrag_nk(sW2 + nt * 16 * 256, 256, kt, lane), t2);
    }
    int j = nt * 16 + col;
    #pragma unroll
    for (int g = 0; g < 8; ++g) {
      int row = g + 8 * hi;
      float gi = sGV[(row * 128 + j) * 2 + 0];
      float vc = sGV[(row * 128 + j) * 2 + 1];
      out[(bBase + row) * 256 + 128 + j] = (2.f * vc * t2[g] - t1[g]) * gi;
    }
  }

  // ---- out[:, :128] = v (straight f32 copy from input) ----
  for (int i = sub * 32 + lane; i < 2048; i += 64) {
    int r = i >> 7, j = i & 127;
    out[(bBase + r) * 256 + j] = in[(bBase + r) * 256 + 128 + j];
  }
}

extern "C" void kernel_launch(void* const* d_in, const int* in_sizes, int n_in,
                              void* d_out, int out_size, void* d_ws, size_t ws_size,
                              hipStream_t stream) {
  (void)in_sizes; (void)n_in; (void)out_size; (void)ws_size;
  const float* in = (const float*)d_in[1];   // input_ (4096 x 256)
  const float* W1 = (const float*)d_in[2];   // 256 x 128
  const float* b1 = (const float*)d_in[3];   // 256
  const float* W2 = (const float*)d_in[4];   // 128 x 256
  const float* b2 = (const float*)d_in[5];   // 128
  float* out = (float*)d_out;
  _Float16* wsh = (_Float16*)d_ws;           // 131072 f16 = 256KB scratch

  // Allow >48KB dynamic LDS (deterministic; safe to call every launch).
  (void)hipFuncSetAttribute((const void*)conn_fused,
                            hipFuncAttributeMaxDynamicSharedMemorySize,
                            (int)SMEM_TOTAL);

  convert_weights<<<256, 256, 0, stream>>>(W1, W2, wsh);
  conn_fused<<<128, 128, SMEM_TOTAL, stream>>>(in, wsh, b1, b2, out);
}
